// FlowNet_70729521431133
// MI455X (gfx1250) — matrix operand order repaired and checked by reading, not compiled, not run on hardware
//
#include <hip/hip_runtime.h>
#include <hip/hip_bf16.h>
#include <math.h>

// ---------------------------------------------------------------------------
// Problem constants (B=2, C=32, D=H=W=64)
// ---------------------------------------------------------------------------
constexpr int CH   = 32;
constexpr int DHW  = 64 * 64 * 64;   // 262144
constexpr int NVOX = 2 * DHW;        // 524288 voxels (B * D*H*W)
constexpr int NEL3 = 2 * 3 * DHW;    // 1572864 flow elements
constexpr int NBLK = NVOX / 256;     // 2048 reduction blocks
constexpr float EPS = 1e-6f;

typedef __attribute__((ext_vector_type(16))) _Float16 v16h;
typedef __attribute__((ext_vector_type(8)))  float    v8f;

// ---------------------------------------------------------------------------
// Helpers
// ---------------------------------------------------------------------------
__device__ inline float smoothl1f(float d) {
    float a = fabsf(d);
    return (a < 1.0f) ? 0.5f * d * d : a - 0.5f;
}

// deterministic block tree-reduction (blockDim.x == 256)
__device__ inline float blockSum256(float v, float* s) {
    int t = threadIdx.x;
    s[t] = v;
    __syncthreads();
    #pragma unroll
    for (int off = 128; off > 0; off >>= 1) {
        if (t < off) s[t] += s[t + off];
        __syncthreads();
    }
    float r = s[0];
    __syncthreads();
    return r;
}

// trilinear gather of a 3-channel field (B,3,D,H,W), reference backwarp rules
__device__ inline void trilinear3(const float* __restrict__ fld, int b,
                                  float xs, float ys, float zs, float out[3]) {
    float x0 = floorf(xs), y0 = floorf(ys), z0 = floorf(zs);
    out[0] = out[1] = out[2] = 0.f;
    const float* base = fld + (size_t)b * 3 * DHW;
    #pragma unroll
    for (int k = 0; k < 8; ++k) {
        float zi = z0 + (float)(k >> 2);
        float yi = y0 + (float)((k >> 1) & 1);
        float xi = x0 + (float)(k & 1);
        float w = (1.f - fabsf(zs - zi)) * (1.f - fabsf(ys - yi)) * (1.f - fabsf(xs - xi));
        bool ok = (zi >= 0.f) && (zi < 64.f) && (yi >= 0.f) && (yi < 64.f) &&
                  (xi >= 0.f) && (xi < 64.f);
        float zc = fminf(fmaxf(zi, 0.f), 63.f);
        float yc = fminf(fmaxf(yi, 0.f), 63.f);
        float xc = fminf(fmaxf(xi, 0.f), 63.f);
        int idx = (((int)zc * 64 + (int)yc) * 64) + (int)xc;
        float wv = ok ? w : 0.f;
        out[0] = fmaf(base[idx],            wv, out[0]);
        out[1] = fmaf(base[DHW + idx],      wv, out[1]);
        out[2] = fmaf(base[2 * DHW + idx],  wv, out[2]);
    }
}

// ---------------------------------------------------------------------------
// Kernel 1: fused corr3d + flow_head.
// Wave32: each lane owns one voxel (32 voxels/wave). corr = 27 dots over C=32
// with VALU FMAs; the W1 (64x27 pad 32) matmul runs on V_WMMA_F32_16X16X32_F16
// (4 M-tiles x 2 N-tiles). Both W1 and the cc tile are staged in LDS already
// in WMMA *fragment order*, so every v16h fragment is one contiguous 32-byte
// LDS access (2x ds_load_b128) instead of 16 scattered ds_load_u16.
// W2 (3x64) is per-lane FMA + cross-half shuffle reduce.
// ---------------------------------------------------------------------------
__global__ __launch_bounds__(256)
void corr_flow_kernel(const float* __restrict__ f0, const float* __restrict__ f1,
                      const float* __restrict__ W1, const float* __restrict__ b1,
                      const float* __restrict__ W2, const float* __restrict__ b2,
                      float* __restrict__ flow_out) {
    // A fragments of W1 in fragment order: w1f[t][lane][e]  (4 KB)
    __shared__ __attribute__((aligned(32))) _Float16 w1f[4][32][16];
    __shared__ float b1s[64];
    __shared__ float w2s[3 * 64];
    __shared__ float b2s[4];
    // cc tile in B-fragment order: ccf[wave][nt][reader_lane][e]  (16 KB)
    __shared__ __attribute__((aligned(32))) _Float16 ccf[8][2][32][16];

    const int tid  = threadIdx.x;
    const int lane = tid & 31;
    const int wave = tid >> 5;

    // ---- stage W1 in A-fragment layout -------------------------------------
    // A order used here (contiguous k, same as B): lane<16: K(e)= e
    //                            lane>=16: K(e)= e+16 ; M = lane&15
    for (int i = tid; i < 4 * 32 * 16; i += 256) {
        int e  = i & 15;
        int ln = (i >> 4) & 31;
        int t  = i >> 9;
        int m  = t * 16 + (ln & 15);
        int k  = (ln < 16) ? e : e + 16;
        (&w1f[0][0][0])[i] = (_Float16)((k < 27) ? W1[m * 27 + k] : 0.0f);
    }
    if (tid < 64)  b1s[tid] = b1[tid];
    if (tid < 192) w2s[tid] = W2[tid];
    if (tid < 3)   b2s[tid] = b2[tid];
    __syncthreads();

    const int vbase = (blockIdx.x * 8 + wave) * 32;
    const int v  = vbase + lane;
    const int b  = v >> 18;                // / DHW
    const int sp = v & (DHW - 1);
    const int z  = sp >> 12;
    const int y  = (sp >> 6) & 63;
    const int x  = sp & 63;

    // ---- corr3d: 27 accumulators -------------------------------------------
    unsigned vmask = 0;
    #pragma unroll
    for (int o = 0; o < 27; ++o) {
        int dz = o / 9 - 1, dy = (o / 3) % 3 - 1, dx = o % 3 - 1;
        bool ok = ((unsigned)(z + dz) < 64u) && ((unsigned)(y + dy) < 64u) &&
                  ((unsigned)(x + dx) < 64u);
        vmask |= ok ? (1u << o) : 0u;
    }

    const float* f0b = f0 + (size_t)b * CH * DHW + sp;
    const float* f1b = f1 + (size_t)b * CH * DHW + sp;

    float acc[27];
    #pragma unroll
    for (int o = 0; o < 27; ++o) acc[o] = 0.f;

    for (int c = 0; c < CH; ++c) {
        const float* p = f1b + (size_t)c * DHW;
        if (c + 1 < CH) {
            __builtin_prefetch(p + DHW, 0, 1);          // global_prefetch_b8
            __builtin_prefetch(f0b + (size_t)(c + 1) * DHW, 0, 1);
        }
        float a = f0b[(size_t)c * DHW];
        #pragma unroll
        for (int o = 0; o < 27; ++o) {
            int dz = o / 9 - 1, dy = (o / 3) % 3 - 1, dx = o % 3 - 1;
            int delta = dz * 4096 + dy * 64 + dx;
            float bv = ((vmask >> o) & 1u) ? p[delta] : 0.f;
            acc[o] = fmaf(a, bv, acc[o]);
        }
    }

    // ---- flow_head front: relu + L2 normalize over 27 channels -------------
    float ss = 0.f;
    #pragma unroll
    for (int o = 0; o < 27; ++o) {
        float r = acc[o] > 0.f ? acc[o] : 0.f;
        acc[o] = r;
        ss = fmaf(r, r, ss);
    }
    float inv = 1.0f / (sqrtf(ss) + EPS);

    // ---- scatter cc into B-fragment order (2 contiguous 32B stores) --------
    // B order used here (32x16 f16, contiguous k as in A): reader lane L: col = L&15, K(e) = e + (L>=16?16:0)
    // This lane owns column n = lane (tile nt = lane>>4, col-in-tile cl).
    {
        v16h lowv, highv;
        #pragma unroll
        for (int e = 0; e < 16; ++e) {
            float lo = (e < 27) ? acc[e] * inv : 0.f;       // K = e
            int ke = 16 + e;                                 // K = 16+e
            float hi = (ke < 27) ? acc[ke] * inv : 0.f;
            lowv[e]  = (_Float16)lo;
            highv[e] = (_Float16)hi;
        }
        int nt = lane >> 4, cl = lane & 15;
        *(v16h*)&ccf[wave][nt][cl][0]      = lowv;   // K 0..15  -> reader cl
        *(v16h*)&ccf[wave][nt][cl + 16][0] = highv;  // K 16..31 -> reader cl+16
    }
    __syncthreads();

    // ---- WMMA: 4 M-tiles x 2 N-tiles ---------------------------------------
    const bool hi = lane >= 16;
    v16h afrag[4];
    #pragma unroll
    for (int t = 0; t < 4; ++t)
        afrag[t] = *(const v16h*)&w1f[t][lane][0];   // 2x ds_load_b128

    #pragma unroll
    for (int nt = 0; nt < 2; ++nt) {
        v16h bfrag = *(const v16h*)&ccf[wave][nt][lane][0];  // 2x ds_load_b128

        float part0 = 0.f, part1 = 0.f, part2 = 0.f;
        #pragma unroll
        for (int t = 0; t < 4; ++t) {
            v8f d = {};
            d = __builtin_amdgcn_wmma_f32_16x16x32_f16(
                    false, afrag[t], false, bfrag, (short)0, d, false, false);
            #pragma unroll
            for (int r = 0; r < 8; ++r) {
                int m = t * 16 + r + (hi ? 8 : 0);
                float hval = d[r] + b1s[m];
                hval = hval > 0.f ? hval : 0.01f * hval;   // leaky relu
                part0 = fmaf(w2s[0 * 64 + m], hval, part0);
                part1 = fmaf(w2s[1 * 64 + m], hval, part1);
                part2 = fmaf(w2s[2 * 64 + m], hval, part2);
            }
        }
        // combine the two lane-halves holding complementary M rows
        part0 += __shfl_xor(part0, 16, 32);
        part1 += __shfl_xor(part1, 16, 32);
        part2 += __shfl_xor(part2, 16, 32);

        if (lane < 16) {
            int vv = vbase + nt * 16 + lane;
            int bb = vv >> 18;
            int sv = vv & (DHW - 1);
            float* fo = flow_out + (size_t)bb * 3 * DHW + sv;
            fo[0]        = part0 + b2s[0];
            fo[DHW]      = part1 + b2s[1];
            fo[2 * DHW]  = part2 + b2s[2];
        }
    }
}

// ---------------------------------------------------------------------------
// Kernel 2: hinge loss + balanced L1/L2 flow losses (terms 0..7)
// ---------------------------------------------------------------------------
__global__ __launch_bounds__(256)
void voxel_loss_kernel(const float* __restrict__ f0, const float* __restrict__ f1,
                       const float* __restrict__ flow_g, const float* __restrict__ mask_g,
                       const float* __restrict__ flow_f, float* __restrict__ partials) {
    __shared__ float red[256];
    const int v  = blockIdx.x * 256 + threadIdx.x;
    const int b  = v >> 18;
    const int sp = v & (DHW - 1);

    // hinge_vox = relu(0.2 - sqrt(sum_c (f1-f0)^2 + EPS))
    float ssq = EPS;
    const float* p0 = f0 + (size_t)b * CH * DHW + sp;
    const float* p1 = f1 + (size_t)b * CH * DHW + sp;
    for (int c = 0; c < CH; ++c) {
        float d = p1[(size_t)c * DHW] - p0[(size_t)c * DHW];
        ssq = fmaf(d, d, ssq);
    }
    float hv = 0.2f - sqrtf(ssq);
    hv = hv > 0.f ? hv : 0.f;

    const float* pg = flow_g + (size_t)b * 3 * DHW + sp;
    float g0 = pg[0], g1 = pg[DHW], g2 = pg[2 * DHW];
    float gabs = fabsf(g0) + fabsf(g1) + fabsf(g2);
    float hm = (gabs > 1.0f) ? 1.f : 0.f;

    const float* pf = flow_f + (size_t)b * 3 * DHW + sp;
    float d0 = pf[0] - g0, d1 = pf[DHW] - g1, d2 = pf[2 * DHW] - g2;
    float l1 = (smoothl1f(d0) + smoothl1f(d1) + smoothl1f(d2)) * (1.f / 3.f);
    float l2 = (d0 * d0 + d1 * d1 + d2 * d2) * (1.f / 3.f);

    float mg  = mask_g[v];
    float ind = (gabs > 0.01f) ? 1.f : 0.f;
    float nz  = ind * mg;
    float yz  = (1.f - nz) * mg;

    float terms[8] = { hv * hm, hm, l1 * nz, l2 * nz, nz, l1 * yz, l2 * yz, yz };
    #pragma unroll
    for (int t = 0; t < 8; ++t) {
        float s = blockSum256(terms[t], red);
        if (threadIdx.x == 0) partials[t * NBLK + blockIdx.x] = s;
    }
}

// ---------------------------------------------------------------------------
// Kernel 3: warp loss — backwarp(feat1, flow_forw) vs feat0 (terms 8..9)
// ---------------------------------------------------------------------------
__global__ __launch_bounds__(256)
void warp_loss_kernel(const float* __restrict__ f0, const float* __restrict__ f1,
                      const float* __restrict__ flow_f, float* __restrict__ partials) {
    __shared__ float red[256];
    const int v  = blockIdx.x * 256 + threadIdx.x;
    const int b  = v >> 18;
    const int sp = v & (DHW - 1);
    const int z  = sp >> 12, y = (sp >> 6) & 63, x = sp & 63;

    const float* pf = flow_f + (size_t)b * 3 * DHW + sp;
    float xs = (float)x + pf[0];
    float ys = (float)y + pf[DHW];
    float zs = (float)z + pf[2 * DHW];
    float x0 = floorf(xs), y0 = floorf(ys), z0 = floorf(zs);

    float wc[8]; int ic[8]; float wsum = 0.f;
    #pragma unroll
    for (int k = 0; k < 8; ++k) {
        float zi = z0 + (float)(k >> 2);
        float yi = y0 + (float)((k >> 1) & 1);
        float xi = x0 + (float)(k & 1);
        float w = (1.f - fabsf(zs - zi)) * (1.f - fabsf(ys - yi)) * (1.f - fabsf(xs - xi));
        bool ok = (zi >= 0.f) && (zi < 64.f) && (yi >= 0.f) && (yi < 64.f) &&
                  (xi >= 0.f) && (xi < 64.f);
        float zc = fminf(fmaxf(zi, 0.f), 63.f);
        float yc = fminf(fmaxf(yi, 0.f), 63.f);
        float xc = fminf(fmaxf(xi, 0.f), 63.f);
        ic[k] = (((int)zc * 64 + (int)yc) * 64) + (int)xc;
        wc[k] = ok ? w : 0.f;
        wsum += wc[k];
    }

    float ssq = EPS;
    const float* f1b = f1 + (size_t)b * CH * DHW;
    const float* f0b = f0 + (size_t)b * CH * DHW + sp;
    for (int c = 0; c < CH; ++c) {
        const float* pc = f1b + (size_t)c * DHW;
        float g = 0.f;
        #pragma unroll
        for (int k = 0; k < 8; ++k) g = fmaf(pc[ic[k]], wc[k], g);
        float d = g - f0b[(size_t)c * DHW];
        ssq = fmaf(d, d, ssq);
    }
    float l2 = sqrtf(ssq);
    float wm = wsum * wsum;        // valid1 * valid0 (both identical)

    float s0 = blockSum256(l2 * wm, red);
    if (threadIdx.x == 0) partials[8 * NBLK + blockIdx.x] = s0;
    float s1 = blockSum256(wm, red);
    if (threadIdx.x == 0) partials[9 * NBLK + blockIdx.x] = s1;
}

// ---------------------------------------------------------------------------
// Kernel 4: cycle loss, both directions (terms 10..11)
// ---------------------------------------------------------------------------
__global__ __launch_bounds__(256)
void cycle_loss_kernel(const float* __restrict__ flow_f, const float* __restrict__ flow_b,
                       float* __restrict__ partials) {
    __shared__ float red[256];
    const int v  = blockIdx.x * 256 + threadIdx.x;
    const int b  = v >> 18;
    const int sp = v & (DHW - 1);
    const int z  = sp >> 12, y = (sp >> 6) & 63, x = sp & 63;

    const float* pf = flow_f + (size_t)b * 3 * DHW + sp;
    const float* pb = flow_b + (size_t)b * 3 * DHW + sp;
    float ff[3] = { pf[0], pf[DHW], pf[2 * DHW] };
    float fb[3] = { pb[0], pb[DHW], pb[2 * DHW] };

    float al[3];
    trilinear3(flow_b, b, (float)x + ff[0], (float)y + ff[1], (float)z + ff[2], al);
    float cf = smoothl1f(ff[0] + al[0]) + smoothl1f(ff[1] + al[1]) + smoothl1f(ff[2] + al[2]);

    trilinear3(flow_f, b, (float)x + fb[0], (float)y + fb[1], (float)z + fb[2], al);
    float cb = smoothl1f(fb[0] + al[0]) + smoothl1f(fb[1] + al[1]) + smoothl1f(fb[2] + al[2]);

    float s0 = blockSum256(cf, red);
    if (threadIdx.x == 0) partials[10 * NBLK + blockIdx.x] = s0;
    float s1 = blockSum256(cb, red);
    if (threadIdx.x == 0) partials[11 * NBLK + blockIdx.x] = s1;
}

// ---------------------------------------------------------------------------
// Kernel 5: smoothness loss (reference counts dx twice; dz unused) (12..13)
// ---------------------------------------------------------------------------
__global__ __launch_bounds__(256)
void smooth_loss_kernel(const float* __restrict__ flow_f, const float* __restrict__ flow_b,
                        float* __restrict__ partials) {
    __shared__ float red[256];
    const int v  = blockIdx.x * 256 + threadIdx.x;
    const int b  = v >> 18;
    const int sp = v & (DHW - 1);
    const int y  = (sp >> 6) & 63, x = sp & 63;

    float sums[2] = { 0.f, 0.f };
    const float* fields[2] = { flow_f, flow_b };
    #pragma unroll
    for (int f = 0; f < 2; ++f) {
        const float* base = fields[f] + (size_t)b * 3 * DHW + sp;
        #pragma unroll
        for (int o = 0; o < 3; ++o) {
            const float* p = base + (size_t)o * DHW;
            float c  = p[0];
            float dx = (x < 63) ? fabsf(p[1] - c)  : 0.f;
            float dy = (y < 63) ? fabsf(p[64] - c) : 0.f;
            sums[f] += 2.f * dx + dy;   // dx + dy + dx (reference quirk)
        }
    }
    float s0 = blockSum256(sums[0], red);
    if (threadIdx.x == 0) partials[12 * NBLK + blockIdx.x] = s0;
    float s1 = blockSum256(sums[1], red);
    if (threadIdx.x == 0) partials[13 * NBLK + blockIdx.x] = s1;
}

// ---------------------------------------------------------------------------
// Kernel 6: final deterministic reduction + loss assembly -> d_out[0]
// ---------------------------------------------------------------------------
__global__ __launch_bounds__(256)
void finalize_kernel(const float* __restrict__ partials, float* __restrict__ out) {
    __shared__ float red[256];
    float S[14];
    for (int t = 0; t < 14; ++t) {
        float acc = 0.f;
        for (int i = threadIdx.x; i < NBLK; i += 256) acc += partials[t * NBLK + i];
        S[t] = blockSum256(acc, red);
    }
    if (threadIdx.x == 0) {
        float hinge  = S[0] / (S[1] + EPS);
        float l1_bal = 0.5f * (S[2] / (S[4] + EPS) + S[5] / (S[7] + EPS));
        float l2_bal = 0.5f * (S[3] / (S[4] + EPS) + S[6] / (S[7] + EPS));
        float warp   = S[8] / (S[9] + EPS);
        float cycle  = (S[10] + S[11]) * (1.f / (float)NEL3);
        float smooth = 0.5f * (S[12] + S[13]) * (1.f / (float)NEL3);
        out[0] = hinge + cycle + l1_bal + l2_bal + warp + smooth;
    }
}

// ---------------------------------------------------------------------------
// Launch: d_out = [total_loss (1 float), flow_forw (2*3*64^3 floats)]
// d_ws   = [flow_back (NEL3 floats), partials (14*NBLK floats)]
// ---------------------------------------------------------------------------
extern "C" void kernel_launch(void* const* d_in, const int* in_sizes, int n_in,
                              void* d_out, int out_size, void* d_ws, size_t ws_size,
                              hipStream_t stream) {
    (void)in_sizes; (void)n_in; (void)out_size; (void)ws_size;
    const float* feat0  = (const float*)d_in[0];
    const float* feat1  = (const float*)d_in[1];
    const float* flow_g = (const float*)d_in[2];
    const float* mask_g = (const float*)d_in[3];
    const float* W1     = (const float*)d_in[4];
    const float* b1     = (const float*)d_in[5];
    const float* W2     = (const float*)d_in[6];
    const float* b2     = (const float*)d_in[7];

    float* out      = (float*)d_out;
    float* flow_f   = out + 1;
    float* ws       = (float*)d_ws;
    float* flow_b   = ws;
    float* partials = ws + NEL3;

    dim3 blk(256);
    corr_flow_kernel<<<NBLK, blk, 0, stream>>>(feat0, feat1, W1, b1, W2, b2, flow_f);
    corr_flow_kernel<<<NBLK, blk, 0, stream>>>(feat1, feat0, W1, b1, W2, b2, flow_b);
    voxel_loss_kernel<<<NBLK, blk, 0, stream>>>(feat0, feat1, flow_g, mask_g, flow_f, partials);
    warp_loss_kernel<<<NBLK, blk, 0, stream>>>(feat0, feat1, flow_f, partials);
    cycle_loss_kernel<<<NBLK, blk, 0, stream>>>(flow_f, flow_b, partials);
    smooth_loss_kernel<<<NBLK, blk, 0, stream>>>(flow_f, flow_b, partials);
    finalize_kernel<<<1, blk, 0, stream>>>(partials, out);
}
